// GAT_46995532153059
// MI455X (gfx1250) — matrix operand order repaired and checked by reading, not compiled, not run on hardware
//
#include <hip/hip_runtime.h>
#include <stdint.h>

// Problem constants (from reference)
#define DIM   128
#define HID   64
#define OUTD  10
#define EPSV  1e-5f
#define SLOPE 0.2f

typedef __attribute__((ext_vector_type(16))) __bf16 v16bf;
typedef __attribute__((ext_vector_type(8)))  float  v8f;

// ---------------- helpers ----------------
__device__ __forceinline__ float leaky(float v) { return v > 0.0f ? v : SLOPE * v; }

__device__ __forceinline__ unsigned short f2bf_rne(float f) {
    union { float f; unsigned u; } v; v.f = f;
    unsigned r = v.u + 0x7FFFu + ((v.u >> 16) & 1u);
    return (unsigned short)(r >> 16);
}

__device__ __forceinline__ void atomicMaxFloat(float* addr, float val) {
    unsigned* ua = (unsigned*)addr;
    unsigned old = *ua;
    while (true) {
        float f = __uint_as_float(old);
        if (f >= val) break;
        unsigned assumed = old;
        old = atomicCAS(ua, assumed, __float_as_uint(val));
        if (old == assumed) break;
    }
}

// ---------------- tiny utility kernels ----------------
__global__ void k_zero(float* __restrict__ p, long long n) {
    long long i = (long long)blockIdx.x * blockDim.x + threadIdx.x;
    if (i < n) p[i] = 0.0f;
}

// count incoming edges per node (f32)
__global__ void k_count(const long long* __restrict__ ei, float* __restrict__ cnt, int E) {
    int e = blockIdx.x * blockDim.x + threadIdx.x;
    if (e >= E) return;
    int d = (int)ei[(long long)E + e];
    atomicAdd(&cnt[d], 1.0f);
}

// one wave per edge: agg[dst] += x[src]  (128 f32 = 32 lanes x float4)
__global__ void k_scatter_sum(const float* __restrict__ x, const long long* __restrict__ ei,
                              float* __restrict__ agg, int E) {
    long long t = (long long)blockIdx.x * blockDim.x + threadIdx.x;
    int e = (int)(t >> 5);
    if (e >= E) return;
    int lane = (int)(t & 31);
    int s = (int)ei[e];
    int d = (int)ei[(long long)E + e];
    float4 v = ((const float4*)(x + (size_t)s * DIM))[lane];
    float* o = agg + (size_t)d * DIM + lane * 4;
    atomicAdd(o + 0, v.x); atomicAdd(o + 1, v.y);
    atomicAdd(o + 2, v.z); atomicAdd(o + 3, v.w);
}

// agg /= max(cnt,1) elementwise
__global__ void k_mean(float* __restrict__ agg, const float* __restrict__ cnt, long long total) {
    long long i = (long long)blockIdx.x * blockDim.x + threadIdx.x;
    if (i >= total) return;
    float c = cnt[i >> 7];
    agg[i] *= 1.0f / fmaxf(c, 1.0f);
}

// cnt[i] = rsqrt(cnt[i] + 1)   (GCN: deg includes self loop, always >= 1)
__global__ void k_dinv(float* __restrict__ cnt, int M) {
    int i = blockIdx.x * blockDim.x + threadIdx.x;
    if (i < M) cnt[i] = rsqrtf(cnt[i] + 1.0f);
}

// out[i,c] = xw[i,c]*dinv[i]^2 + bias[c]   (GCN self-loop init)
__global__ void k_gcn_selfinit(const float* __restrict__ xw, const float* __restrict__ dinv,
                               const float* __restrict__ bias, float* __restrict__ out,
                               long long total) {
    long long i = (long long)blockIdx.x * blockDim.x + threadIdx.x;
    if (i >= total) return;
    long long row = i >> 7; int col = (int)(i & 127);
    float dv = dinv[row];
    out[i] = xw[i] * dv * dv + bias[col];
}

// one wave per edge: out[d] += xw[s] * dinv[s]*dinv[d]
__global__ void k_gcn_edge(const float* __restrict__ xw, const float* __restrict__ dinv,
                           const long long* __restrict__ ei, float* __restrict__ out, int E) {
    long long t = (long long)blockIdx.x * blockDim.x + threadIdx.x;
    int e = (int)(t >> 5);
    if (e >= E) return;
    int lane = (int)(t & 31);
    int s = (int)ei[e];
    int d = (int)ei[(long long)E + e];
    float cf = dinv[s] * dinv[d];
    float4 v = ((const float4*)(xw + (size_t)s * DIM))[lane];
    float* o = out + (size_t)d * DIM + lane * 4;
    atomicAdd(o + 0, cf * v.x); atomicAdd(o + 1, cf * v.y);
    atomicAdd(o + 2, cf * v.z); atomicAdd(o + 3, cf * v.w);
}

// ---------------- BatchNorm ----------------
__global__ void k_bn_stats(const float* __restrict__ X, float* __restrict__ sum,
                           float* __restrict__ sumsq, int M) {
    int col = threadIdx.x;            // 128 threads
    int r0 = blockIdx.x * 128;
    float s = 0.0f, q = 0.0f;
    for (int i = 0; i < 128; i++) {
        int r = r0 + i;
        if (r < M) {
            float v = X[(size_t)r * DIM + col];
            s += v; q += v * v;
        }
    }
    atomicAdd(&sum[col], s);
    atomicAdd(&sumsq[col], q);
}

__global__ void k_bn_apply(float* __restrict__ X, const float* __restrict__ sum,
                           const float* __restrict__ sumsq, const float* __restrict__ g,
                           const float* __restrict__ b, int M, long long total) {
    long long i = (long long)blockIdx.x * blockDim.x + threadIdx.x;
    if (i >= total) return;
    int col = (int)(i & 127);
    float inv = 1.0f / (float)M;
    float mu  = sum[col] * inv;
    float var = sumsq[col] * inv - mu * mu;
    float y = (X[i] - mu) * rsqrtf(var + EPSV) * g[col] + b[col];
    X[i] = fmaxf(y, 0.0f);
}

// ---------------- GAT ----------------
__global__ void k_gat_prep(const float* __restrict__ xw, const float* __restrict__ att_s,
                           const float* __restrict__ att_d, float* __restrict__ as,
                           float* __restrict__ ad, float* __restrict__ mx, int M) {
    int i = blockIdx.x * blockDim.x + threadIdx.x;
    if (i >= M) return;
    const float4* row = (const float4*)(xw + (size_t)i * DIM);
    const float4* vs  = (const float4*)att_s;
    const float4* vd  = (const float4*)att_d;
    float s = 0.0f, d = 0.0f;
    #pragma unroll 8
    for (int j = 0; j < 32; j++) {
        float4 r = row[j], a = vs[j], b = vd[j];
        s += r.x * a.x + r.y * a.y + r.z * a.z + r.w * a.w;
        d += r.x * b.x + r.y * b.y + r.z * b.z + r.w * b.w;
    }
    as[i] = s; ad[i] = d;
    mx[i] = leaky(s + d);   // self-loop attention score seeds the max
}

__global__ void k_gat_max(const float* __restrict__ as, const float* __restrict__ ad,
                          const long long* __restrict__ ei, float* __restrict__ mx, int E) {
    int e = blockIdx.x * blockDim.x + threadIdx.x;
    if (e >= E) return;
    int s = (int)ei[e];
    int d = (int)ei[(long long)E + e];
    atomicMaxFloat(&mx[d], leaky(as[s] + ad[d]));
}

// one wave per node: z[i]=exp(e_self-m); num[i]=z_self*xw[i]
__global__ void k_gat_selfinit(const float* __restrict__ xw, const float* __restrict__ as,
                               const float* __restrict__ ad, const float* __restrict__ mx,
                               float* __restrict__ z, float* __restrict__ num, int M) {
    long long t = (long long)blockIdx.x * blockDim.x + threadIdx.x;
    int i = (int)(t >> 5);
    if (i >= M) return;
    int lane = (int)(t & 31);
    float w = __expf(leaky(as[i] + ad[i]) - mx[i]);
    if (lane == 0) z[i] = w;
    float4 v = ((const float4*)(xw + (size_t)i * DIM))[lane];
    float4 o = make_float4(v.x * w, v.y * w, v.z * w, v.w * w);
    ((float4*)(num + (size_t)i * DIM))[lane] = o;
}

__global__ void k_gat_edge(const float* __restrict__ xw, const float* __restrict__ as,
                           const float* __restrict__ ad, const float* __restrict__ mx,
                           const long long* __restrict__ ei, float* __restrict__ z,
                           float* __restrict__ num, int E) {
    long long t = (long long)blockIdx.x * blockDim.x + threadIdx.x;
    int e = (int)(t >> 5);
    if (e >= E) return;
    int lane = (int)(t & 31);
    int s = (int)ei[e];
    int d = (int)ei[(long long)E + e];
    float ee = __expf(leaky(as[s] + ad[d]) - mx[d]);
    if (lane == 0) atomicAdd(&z[d], ee);
    float4 v = ((const float4*)(xw + (size_t)s * DIM))[lane];
    float* o = num + (size_t)d * DIM + lane * 4;
    atomicAdd(o + 0, ee * v.x); atomicAdd(o + 1, ee * v.y);
    atomicAdd(o + 2, ee * v.z); atomicAdd(o + 3, ee * v.w);
}

__global__ void k_gat_final(float* __restrict__ num, const float* __restrict__ z,
                            const float* __restrict__ bias, long long total) {
    long long i = (long long)blockIdx.x * blockDim.x + threadIdx.x;
    if (i >= total) return;
    long long row = i >> 7; int col = (int)(i & 127);
    num[i] = num[i] / z[row] + bias[col];
}

// ---------------- weight packing (B-fragment layout, contiguous k, bf16) ----------------
// B for v_wmma_f32_16x16x32_bf16 (32x16), in this code's contiguous k order:
//   lane l: n = nt*16 + (l&15), kbase = kt*32 + (l>>4)*16
//   dword v (0..7) packs {W[kbase+2v][n], W[kbase+2v+1][n]}
__global__ void k_pack_weight(const float* __restrict__ W, uint32_t* __restrict__ out,
                              int Ncols, int KT, int NT) {
    int t = blockIdx.x * blockDim.x + threadIdx.x;
    int total = KT * NT * 32;
    if (t >= total) return;
    int lane = t & 31;
    int nt = (t >> 5) % NT;
    int kt = (t >> 5) / NT;
    int n = nt * 16 + (lane & 15);
    int kbase = kt * 32 + (lane >> 4) * 16;
    uint32_t* o = out + (size_t)t * 8;
    #pragma unroll
    for (int v = 0; v < 8; v++) {
        unsigned short lo = f2bf_rne(W[(size_t)(kbase + 2 * v) * Ncols + n]);
        unsigned short hi = f2bf_rne(W[(size_t)(kbase + 2 * v + 1) * Ncols + n]);
        o[v] = (uint32_t)lo | ((uint32_t)hi << 16);
    }
}

// ---------------- WMMA GEMM ----------------
// C[M x NT*16] = A[M x KT*32] @ Bp (+ A2 @ Bp2 if DUAL) + bias, optional ReLU.
// 256 threads = 8 waves; wave w owns rows [blk*128 + w*16, +16), all NT col tiles.
template <int KT, int NT, bool DUAL>
__global__ __launch_bounds__(256)
void k_gemm_wmma(const float* __restrict__ A, const uint32_t* __restrict__ Bp,
                 const float* __restrict__ A2, const uint32_t* __restrict__ Bp2,
                 const float* __restrict__ bias, float* __restrict__ C,
                 int M, int relu) {
    __shared__ uint32_t sB[(DUAL ? 2 : 1) * KT * NT * 256];
    const int Kdim = KT * 32;
    const int Ncols = NT * 16;
    const int nB = KT * NT * 256;

    int tid = threadIdx.x;
    for (int i = tid; i < nB; i += 256) sB[i] = Bp[i];
    if (DUAL) for (int i = tid; i < nB; i += 256) sB[nB + i] = Bp2[i];
    __syncthreads();

    int wave = tid >> 5;
    int lane = tid & 31;
    int rowA = blockIdx.x * 128 + wave * 16 + (lane & 15);
    int rA = rowA < M ? rowA : (M - 1);       // clamp loads; stores are guarded
    int khalf = lane >> 4;

    v8f acc[NT];
    #pragma unroll
    for (int nt = 0; nt < NT; nt++) acc[nt] = {};

    for (int kt = 0; kt < KT; kt++) {
        // A fragment: lane&15 = M row, khalf picks K 0-15 / 16-31 (contiguous)
        const float* ap = A + (size_t)rA * Kdim + kt * 32 + khalf * 16;
        v16bf afrag;
        #pragma unroll
        for (int j = 0; j < 8; j++) {
            afrag[j]     = (__bf16)ap[j];
            afrag[8 + j] = (__bf16)ap[8 + j];
        }
        v16bf a2frag;
        if (DUAL) {
            const float* ap2 = A2 + (size_t)rA * Kdim + kt * 32 + khalf * 16;
            #pragma unroll
            for (int j = 0; j < 8; j++) {
                a2frag[j]     = (__bf16)ap2[j];
                a2frag[8 + j] = (__bf16)ap2[8 + j];
            }
        }
        #pragma unroll
        for (int nt = 0; nt < NT; nt++) {
            union { uint32_t u[8]; v16bf v; } b;
            const uint32_t* bp = &sB[((kt * NT + nt) * 32 + lane) * 8];
            #pragma unroll
            for (int j = 0; j < 8; j++) b.u[j] = bp[j];
            acc[nt] = __builtin_amdgcn_wmma_f32_16x16x32_bf16(
                false, afrag, false, b.v, (short)0, acc[nt], false, false);
            if (DUAL) {
                union { uint32_t u[8]; v16bf v; } b2;
                const uint32_t* bp2 = &sB[nB + ((kt * NT + nt) * 32 + lane) * 8];
                #pragma unroll
                for (int j = 0; j < 8; j++) b2.u[j] = bp2[j];
                acc[nt] = __builtin_amdgcn_wmma_f32_16x16x32_bf16(
                    false, a2frag, false, b2.v, (short)0, acc[nt], false, false);
            }
        }
    }

    // C fragment: n = lane&15; row = base + (lane>>4)*8 + vgpr
    int rbase = blockIdx.x * 128 + wave * 16 + khalf * 8;
    int ncol = lane & 15;
    #pragma unroll
    for (int nt = 0; nt < NT; nt++) {
        int n = nt * 16 + ncol;
        float bval = bias ? bias[n] : 0.0f;
        #pragma unroll
        for (int v = 0; v < 8; v++) {
            int r = rbase + v;
            if (r < M) {
                float val = acc[nt][v] + bval;
                if (relu) val = fmaxf(val, 0.0f);
                C[(size_t)r * Ncols + n] = val;
            }
        }
    }
}

// ---------------- MLP tail: logits = hid @ W2 + b2, then log_softmax ----------------
__global__ void k_head(const float* __restrict__ Hid, const float* __restrict__ W2,
                       const float* __restrict__ b2, float* __restrict__ out, int M) {
    int i = blockIdx.x * blockDim.x + threadIdx.x;
    if (i >= M) return;
    const float* h = Hid + (size_t)i * HID;
    float lg[OUTD];
    #pragma unroll
    for (int o = 0; o < OUTD; o++) lg[o] = b2[o];
    for (int k = 0; k < HID; k++) {
        float hv = h[k];
        const float* w = W2 + (size_t)k * OUTD;
        #pragma unroll
        for (int o = 0; o < OUTD; o++) lg[o] += hv * w[o];
    }
    float m = lg[0];
    #pragma unroll
    for (int o = 1; o < OUTD; o++) m = fmaxf(m, lg[o]);
    float s = 0.0f;
    #pragma unroll
    for (int o = 0; o < OUTD; o++) s += __expf(lg[o] - m);
    float ls = __logf(s);
    float* op = out + (size_t)i * OUTD;
    #pragma unroll
    for (int o = 0; o < OUTD; o++) op[o] = lg[o] - m - ls;
}

// ---------------- host launch ----------------
extern "C" void kernel_launch(void* const* d_in, const int* in_sizes, int n_in,
                              void* d_out, int out_size, void* d_ws, size_t ws_size,
                              hipStream_t stream) {
    const float*     x     = (const float*)d_in[0];
    const long long* ei    = (const long long*)d_in[1];
    const float* sage_Wl   = (const float*)d_in[2];
    const float* sage_Wr   = (const float*)d_in[3];
    const float* sage_b    = (const float*)d_in[4];
    const float* gcn_W     = (const float*)d_in[5];
    const float* gcn_b     = (const float*)d_in[6];
    const float* gat_W     = (const float*)d_in[7];
    const float* gat_as    = (const float*)d_in[8];
    const float* gat_ad    = (const float*)d_in[9];
    const float* gat_b     = (const float*)d_in[10];
    const float* bn1_g = (const float*)d_in[11]; const float* bn1_b = (const float*)d_in[12];
    const float* bn2_g = (const float*)d_in[13]; const float* bn2_b = (const float*)d_in[14];
    const float* bn3_g = (const float*)d_in[15]; const float* bn3_b = (const float*)d_in[16];
    const float* lin1_W = (const float*)d_in[17]; const float* lin1_b = (const float*)d_in[18];
    const float* lin2_W = (const float*)d_in[19]; const float* lin2_b = (const float*)d_in[20];

    const int N = in_sizes[0] / DIM;
    const int E = in_sizes[1] / 2;
    const long long ND = (long long)N * DIM;

    // workspace carve-up
    float* B0   = (float*)d_ws;         // N*D
    float* B1   = B0 + ND;              // N*D
    float* B2   = B1 + ND;              // N*D
    float* cnt  = B2 + ND;              // N  (later: dinv, in place)
    float* a_s  = cnt + N;              // N
    float* a_d  = a_s + N;              // N
    float* mbuf = a_d + N;              // N
    float* zbuf = mbuf + N;             // N
    float* csum = zbuf + N;             // 128
    float* csq  = csum + DIM;           // 128
    uint32_t* pWl  = (uint32_t*)(csq + DIM);  // 4*8*32*8 = 8192 dwords each
    uint32_t* pWr  = pWl + 8192;
    uint32_t* pGcn = pWr + 8192;
    uint32_t* pGat = pGcn + 8192;
    uint32_t* pL1  = pGat + 8192;             // 4*4*32*8 = 4096 dwords

    const int TB = 256;
    auto blk  = [&](long long n) { return (unsigned)((n + TB - 1) / TB); };
    const unsigned gElem  = blk(ND);
    const unsigned gEdgeW = blk((long long)E * 32);
    const unsigned gNodeW = blk((long long)N * 32);
    const unsigned gGemm  = (unsigned)((N + 127) / 128);

    // pack weights into bf16 B fragments (contiguous k order)
    k_pack_weight<<<blk(4 * 8 * 32), TB, 0, stream>>>(sage_Wl, pWl, DIM, 4, 8);
    k_pack_weight<<<blk(4 * 8 * 32), TB, 0, stream>>>(sage_Wr, pWr, DIM, 4, 8);
    k_pack_weight<<<blk(4 * 8 * 32), TB, 0, stream>>>(gcn_W,   pGcn, DIM, 4, 8);
    k_pack_weight<<<blk(4 * 8 * 32), TB, 0, stream>>>(gat_W,   pGat, DIM, 4, 8);
    k_pack_weight<<<blk(4 * 4 * 32), TB, 0, stream>>>(lin1_W,  pL1,  HID, 4, 4);

    // ---- SAGE ----
    k_zero<<<gElem, TB, 0, stream>>>(B0, ND);
    k_zero<<<blk(N), TB, 0, stream>>>(cnt, N);
    k_count<<<blk(E), TB, 0, stream>>>(ei, cnt, E);
    k_scatter_sum<<<gEdgeW, TB, 0, stream>>>(x, ei, B0, E);
    k_mean<<<gElem, TB, 0, stream>>>(B0, cnt, ND);
    k_gemm_wmma<4, 8, true><<<gGemm, TB, 0, stream>>>(B0, pWl, x, pWr, sage_b, B1, N, 0);
    k_zero<<<1, TB, 0, stream>>>(csum, 256);
    k_bn_stats<<<gGemm, DIM, 0, stream>>>(B1, csum, csq, N);
    k_bn_apply<<<gElem, TB, 0, stream>>>(B1, csum, csq, bn1_g, bn1_b, N, ND);

    // ---- GCN ----
    k_dinv<<<blk(N), TB, 0, stream>>>(cnt, N);   // cnt -> dinv in place
    k_gemm_wmma<4, 8, false><<<gGemm, TB, 0, stream>>>(B1, pGcn, nullptr, nullptr, nullptr, B2, N, 0);
    k_gcn_selfinit<<<gElem, TB, 0, stream>>>(B2, cnt, gcn_b, B0, ND);
    k_gcn_edge<<<gEdgeW, TB, 0, stream>>>(B2, cnt, ei, B0, E);
    k_zero<<<1, TB, 0, stream>>>(csum, 256);
    k_bn_stats<<<gGemm, DIM, 0, stream>>>(B0, csum, csq, N);
    k_bn_apply<<<gElem, TB, 0, stream>>>(B0, csum, csq, bn2_g, bn2_b, N, ND);

    // ---- GAT ----
    k_gemm_wmma<4, 8, false><<<gGemm, TB, 0, stream>>>(B0, pGat, nullptr, nullptr, nullptr, B1, N, 0);
    k_gat_prep<<<blk(N), TB, 0, stream>>>(B1, gat_as, gat_ad, a_s, a_d, mbuf, N);
    k_gat_max<<<blk(E), TB, 0, stream>>>(a_s, a_d, ei, mbuf, E);
    k_gat_selfinit<<<gNodeW, TB, 0, stream>>>(B1, a_s, a_d, mbuf, zbuf, B0, N);
    k_gat_edge<<<gEdgeW, TB, 0, stream>>>(B1, a_s, a_d, mbuf, ei, zbuf, B0, E);
    k_gat_final<<<gElem, TB, 0, stream>>>(B0, zbuf, gat_b, ND);
    k_zero<<<1, TB, 0, stream>>>(csum, 256);
    k_bn_stats<<<gGemm, DIM, 0, stream>>>(B0, csum, csq, N);
    k_bn_apply<<<gElem, TB, 0, stream>>>(B0, csum, csq, bn3_g, bn3_b, N, ND);

    // ---- MLP head ----
    k_gemm_wmma<4, 4, false><<<gGemm, TB, 0, stream>>>(B0, pL1, nullptr, nullptr, lin1_b, B1, N, 1);
    k_head<<<blk(N), TB, 0, stream>>>(B1, lin2_W, lin2_b, (float*)d_out, N);
}
